// TextCrossAttentionLayer_29523605193400
// MI455X (gfx1250) — matrix operand order repaired and checked by reading, not compiled, not run on hardware
//
#include <hip/hip_runtime.h>
#include <math.h>

typedef __attribute__((ext_vector_type(16))) _Float16 v16h;
typedef __attribute__((ext_vector_type(8)))  float    v8f;

#define D_MODEL 256
#define D_CLIP  768
#define NHEAD   8
#define HEAD_DIM 32
#define Q_LEN   100
#define BATCH   32
#define T_LEN   1203
#define T_PAD   1216   /* 76 * 16 */
#define Q_PAD   112    /* 7 * 16  */
#define QT      7      /* q tiles of 16 */
#define TT      76     /* t tiles of 16 */
#define NIT     38     /* t chunks of 32 */
#define ROWS_QB 3200   /* Q_LEN * BATCH */

// A-matrix f16 16x32 per-lane K mapping (ISA 7.12.2): lanes0-15 VGPR0-3 K0-7,
// VGPR4-7 K16-23; lanes16-31 K8-15 / K24-31.
__device__ __forceinline__ int kmapA(int j, int hi) {
  return (j < 8) ? (hi * 8 + j) : (16 + hi * 8 + (j - 8));
}
// B-matrix f16 32x16 per-lane K mapping: same as A (lanes0-15 K0-7,K16-23; lanes16-31 K8-15,K24-31).
__device__ __forceinline__ int kmapB(int j, int hi) { return (j < 8) ? (hi * 8 + j) : (16 + hi * 8 + (j - 8)); }

__device__ __forceinline__ v8f wmma_f16(v16h a, v16h b, v8f c) {
  return __builtin_amdgcn_wmma_f32_16x16x32_f16(false, a, false, b, (short)0, c,
                                                false, false);
}

__device__ __forceinline__ float4 ld4(const float* p) {
  return *(const float4*)p;
}
__device__ __forceinline__ void put4(v16h& f, int base, float4 v, float msk) {
  f[base + 0] = (_Float16)(v.x * msk);
  f[base + 1] = (_Float16)(v.y * msk);
  f[base + 2] = (_Float16)(v.z * msk);
  f[base + 3] = (_Float16)(v.w * msk);
}

// Build an A fragment from a row pointer: halves j<8 come from cols ks+hi*8+j,
// j>=8 from cols ks+16+hi*8+(j-8); both runs contiguous -> float4 loads.
__device__ __forceinline__ v16h fragA_row(const float* row, int ks, int hi,
                                          float msk) {
  v16h a;
  put4(a, 0,  ld4(row + ks + hi * 8 + 0), msk);
  put4(a, 4,  ld4(row + ks + hi * 8 + 4), msk);
  put4(a, 8,  ld4(row + ks + 16 + hi * 8 + 0), msk);
  put4(a, 12, ld4(row + ks + 16 + hi * 8 + 4), msk);
  return a;
}
// Build a B fragment from a row pointer: same K layout as fragA_row
__device__ __forceinline__ v16h fragB_row(const float* row, int ks, int hi) {
  v16h w;
  put4(w, 0,  ld4(row + ks + hi * 8 + 0), 1.0f);
  put4(w, 4,  ld4(row + ks + hi * 8 + 4), 1.0f);
  put4(w, 8,  ld4(row + ks + 16 + hi * 8 + 0), 1.0f);
  put4(w, 12, ld4(row + ks + 16 + hi * 8 + 4), 1.0f);
  return w;
}

// ---------------- K/V projection: (1203,768) @ W^T(768,256) + bias -----------
__global__ __launch_bounds__(32) void proj_text_kernel(
    const float* __restrict__ text, const float* __restrict__ Wk,
    const float* __restrict__ bk, const float* __restrict__ Wv,
    const float* __restrict__ bv, float* __restrict__ kf,
    float* __restrict__ vf) {
  const int tt = blockIdx.x, nt = blockIdx.y, which = blockIdx.z;
  const float* W = which ? Wv : Wk;
  const float* bias = which ? bv : bk;
  float* outp = which ? vf : kf;
  const int l = threadIdx.x, m = l & 15, hi = l >> 4;
  const int row = tt * 16 + m;      // text row (A)
  const int ncol = nt * 16 + m;     // W row (B lane n)
  const float amask = (row < T_LEN) ? 1.0f : 0.0f;
  const float* arow = text + (size_t)min(row, T_LEN - 1) * D_CLIP;
  const float* wrow = W + (size_t)ncol * D_CLIP;
  v8f acc = {};
  for (int ks = 0; ks < D_CLIP; ks += 32) {
    v16h a = fragA_row(arow, ks, hi, amask);
    v16h w = fragB_row(wrow, ks, hi);
    acc = wmma_f16(a, w, acc);
  }
  const float bb = bias[nt * 16 + (l & 15)];
#pragma unroll
  for (int r = 0; r < 8; ++r) {
    int orow = tt * 16 + hi * 8 + r;
    outp[orow * D_MODEL + nt * 16 + (l & 15)] = acc[r] + bb;
  }
}

// ---------------- Q projection: (tgt+query_pos) @ Wq^T + bq ------------------
__global__ __launch_bounds__(32) void proj_q_kernel(
    const float* __restrict__ tgt, const float* __restrict__ qpos,
    const float* __restrict__ Wq, const float* __restrict__ bq,
    float* __restrict__ qf) {
  const int qt = blockIdx.x, nt = blockIdx.y, b = blockIdx.z;
  const int l = threadIdx.x, m = l & 15, hi = l >> 4;
  const int q = qt * 16 + m;
  const int ncol = nt * 16 + m;
  const float amask = (q < Q_LEN) ? 1.0f : 0.0f;
  const size_t abase = (size_t)(min(q, Q_LEN - 1) * BATCH + b) * D_MODEL;
  const float* arow0 = tgt + abase;
  const float* arow1 = qpos + abase;
  const float* wrow = Wq + (size_t)ncol * D_MODEL;
  v8f acc = {};
  for (int ks = 0; ks < D_MODEL; ks += 32) {
    v16h a;
    {
      float4 x0 = ld4(arow0 + ks + hi * 8 + 0), y0 = ld4(arow1 + ks + hi * 8 + 0);
      float4 x1 = ld4(arow0 + ks + hi * 8 + 4), y1 = ld4(arow1 + ks + hi * 8 + 4);
      float4 x2 = ld4(arow0 + ks + 16 + hi * 8 + 0), y2 = ld4(arow1 + ks + 16 + hi * 8 + 0);
      float4 x3 = ld4(arow0 + ks + 16 + hi * 8 + 4), y3 = ld4(arow1 + ks + 16 + hi * 8 + 4);
      float4 s0 = {x0.x + y0.x, x0.y + y0.y, x0.z + y0.z, x0.w + y0.w};
      float4 s1 = {x1.x + y1.x, x1.y + y1.y, x1.z + y1.z, x1.w + y1.w};
      float4 s2 = {x2.x + y2.x, x2.y + y2.y, x2.z + y2.z, x2.w + y2.w};
      float4 s3 = {x3.x + y3.x, x3.y + y3.y, x3.z + y3.z, x3.w + y3.w};
      put4(a, 0, s0, amask);
      put4(a, 4, s1, amask);
      put4(a, 8, s2, amask);
      put4(a, 12, s3, amask);
    }
    v16h w = fragB_row(wrow, ks, hi);
    acc = wmma_f16(a, w, acc);
  }
  const float bb = bq[nt * 16 + (l & 15)];
#pragma unroll
  for (int r = 0; r < 8; ++r) {
    int qq = qt * 16 + hi * 8 + r;
    qf[(b * Q_PAD + qq) * D_MODEL + nt * 16 + (l & 15)] = acc[r] + bb;
  }
}

// ------------- L2-normalize + pack Q into A-fragment layout ------------------
__global__ __launch_bounds__(32) void pack_q_kernel(
    const float* __restrict__ qf, _Float16* __restrict__ qh) {
  const int qt = blockIdx.x, h = blockIdx.y, b = blockIdx.z;
  const int l = threadIdx.x, m = l & 15, hi = l >> 4;
  const int q = qt * 16 + m;  // always < Q_PAD: buffer padded, load freely
  const float* rowp = qf + (size_t)(b * Q_PAD + q) * D_MODEL + h * HEAD_DIM;
  float vals[32];
  float ss = 0.0f;
#pragma unroll
  for (int i = 0; i < 8; ++i) {
    float4 v = ld4(rowp + i * 4);
    vals[i * 4 + 0] = v.x; vals[i * 4 + 1] = v.y;
    vals[i * 4 + 2] = v.z; vals[i * 4 + 3] = v.w;
    ss += v.x * v.x + v.y * v.y + v.z * v.z + v.w * v.w;
  }
  const float inv =
      (q < Q_LEN) ? (1.0f / fmaxf(sqrtf(ss), 1e-6f)) : 0.0f;
  v16h frag;
#pragma unroll
  for (int j = 0; j < 16; ++j)
    frag[j] = (_Float16)(vals[kmapA(j, hi)] * inv);
  *(v16h*)(qh + (size_t)(((b * NHEAD + h) * QT + qt) * 32 + l) * 16) = frag;
}

// ------------- L2-normalize + pack K into B-fragment layout ------------------
__global__ __launch_bounds__(32) void pack_k_kernel(
    const float* __restrict__ kf, _Float16* __restrict__ kh) {
  const int tt = blockIdx.x, h = blockIdx.y;
  const int l = threadIdx.x, m = l & 15, hi = l >> 4;
  const int t = tt * 16 + m;  // always < T_PAD: buffer padded, load freely
  const float* rowp = kf + (size_t)t * D_MODEL + h * HEAD_DIM;
  float vals[32];
  float ss = 0.0f;
#pragma unroll
  for (int i = 0; i < 8; ++i) {
    float4 v = ld4(rowp + i * 4);
    vals[i * 4 + 0] = v.x; vals[i * 4 + 1] = v.y;
    vals[i * 4 + 2] = v.z; vals[i * 4 + 3] = v.w;
    ss += v.x * v.x + v.y * v.y + v.z * v.z + v.w * v.w;
  }
  const float inv =
      (t < T_LEN) ? (1.0f / fmaxf(sqrtf(ss), 1e-6f)) : 0.0f;
  v16h frag;
#pragma unroll
  for (int j = 0; j < 16; ++j)
    frag[j] = (_Float16)(vals[kmapB(j, hi)] * inv);
  *(v16h*)(kh + (size_t)((h * TT + tt) * 32 + l) * 16) = frag;
}

// ------------- Pack V into B-fragment layout (K = t chunk of 32) -------------
// Padded-t rows carry bias values, but their softmax probability is exactly 0,
// so no masking is needed here.
__global__ __launch_bounds__(32) void pack_v_kernel(
    const float* __restrict__ vf, _Float16* __restrict__ vh) {
  const int it = blockIdx.x, nt = blockIdx.y, h = blockIdx.z;
  const int l = threadIdx.x, n = l & 15, hi = l >> 4;
  v16h frag;
#pragma unroll
  for (int j = 0; j < 16; ++j) {
    int t = it * 32 + ((j < 8) ? (hi * 8 + j) : (16 + hi * 8 + (j - 8)));  // < T_PAD always
    frag[j] = (_Float16)vf[(size_t)t * D_MODEL + h * HEAD_DIM + nt * 16 + n];
  }
  *(v16h*)(vh + (size_t)(((h * NIT + it) * 2 + nt) * 32 + l) * 16) = frag;
}

// ------------- Fused flash attention per (q-tile, head, batch) ---------------
__global__ __launch_bounds__(32) void attn_kernel(
    const _Float16* __restrict__ qh, const _Float16* __restrict__ kh,
    const _Float16* __restrict__ vh, const float* __restrict__ oov,
    const float* __restrict__ logit_scale, float* __restrict__ att_out) {
  __shared__ __align__(16) _Float16 pb[16 * 32];
  const int qt = blockIdx.x, h = blockIdx.y, b = blockIdx.z;
  const int l = threadIdx.x, m = l & 15, hi = l >> 4;
  const float scale = fminf(__expf(logit_scale[0]), 100.0f);
  const float alpha = (float)h * (1.0f / 7.0f);
  const float c1 = scale * (1.0f - alpha);
  const float c2 = alpha;
  const float* oovb = oov + (size_t)b * Q_LEN * T_LEN;

  const v16h aq =
      *(const v16h*)(qh + (size_t)(((b * NHEAD + h) * QT + qt) * 32 + l) * 16);
  v8f o0 = {}, o1 = {};
  float mrow[8], lrow[8];
#pragma unroll
  for (int r = 0; r < 8; ++r) { mrow[r] = -1e30f; lrow[r] = 0.0f; }

  const _Float16* khb = kh + (size_t)(h * TT) * 512;
  const _Float16* vhb = vh + (size_t)(h * NIT) * 1024;

  for (int it = 0; it < NIT; ++it) {
    // prefetch next iteration's fragments while this one computes
    {
      int nx = min(it + 1, NIT - 1);
      __builtin_prefetch(khb + (size_t)(nx * 2) * 512 + l * 16, 0, 3);
      __builtin_prefetch(vhb + (size_t)nx * 1024 + l * 16, 0, 3);
    }
    const v16h bk0 = *(const v16h*)(khb + (size_t)(it * 2 + 0) * 512 + l * 16);
    const v16h bk1 = *(const v16h*)(khb + (size_t)(it * 2 + 1) * 512 + l * 16);
    v8f z = {};
    v8f s0 = wmma_f16(aq, bk0, z);
    v8f s1 = wmma_f16(aq, bk1, z);

    const int t0 = it * 32 + (l & 15);
    const int t1 = t0 + 16;
    const int t0c = min(t0, T_LEN - 1);
    const int t1c = min(t1, T_LEN - 1);
#pragma unroll
    for (int r = 0; r < 8; ++r) {
      int q = qt * 16 + hi * 8 + r;
      int qc = min(q, Q_LEN - 1);
      float mq = (q < Q_LEN) ? 1.0f : 0.0f;
      float ov0 = oovb[qc * T_LEN + t0c] * mq;
      float ov1 = oovb[qc * T_LEN + t1c] * mq;
      s0[r] = (t0 < T_LEN) ? (s0[r] * c1 + ov0 * c2) : -1e30f;
      s1[r] = (t1 < T_LEN) ? (s1[r] * c1 + ov1 * c2) : -1e30f;
    }

    // online softmax: row stats live replicated across each 16-lane half
#pragma unroll
    for (int r = 0; r < 8; ++r) {
      float tm = fmaxf(s0[r], s1[r]);
      tm = fmaxf(tm, __shfl_xor(tm, 1, 32));
      tm = fmaxf(tm, __shfl_xor(tm, 2, 32));
      tm = fmaxf(tm, __shfl_xor(tm, 4, 32));
      tm = fmaxf(tm, __shfl_xor(tm, 8, 32));
      float mnew = fmaxf(mrow[r], tm);
      float corr = __expf(mrow[r] - mnew);
      mrow[r] = mnew;
      float p0 = __expf(s0[r] - mnew);
      float p1 = __expf(s1[r] - mnew);
      s0[r] = p0;
      s1[r] = p1;
      float ps = p0 + p1;
      ps += __shfl_xor(ps, 1, 32);
      ps += __shfl_xor(ps, 2, 32);
      ps += __shfl_xor(ps, 4, 32);
      ps += __shfl_xor(ps, 8, 32);
      lrow[r] = lrow[r] * corr + ps;
      o0[r] *= corr;
      o1[r] *= corr;
    }

    // transpose P (C layout) -> A-fragment layout through LDS
#pragma unroll
    for (int r = 0; r < 8; ++r) {
      int row = hi * 8 + r;
      pb[row * 32 + (l & 15)] = (_Float16)s0[r];
      pb[row * 32 + 16 + (l & 15)] = (_Float16)s1[r];
    }
    __syncthreads();
    v16h pa;
    const _Float16* prow = pb + m * 32 + hi * 8;
#pragma unroll
    for (int j = 0; j < 8; ++j) pa[j] = prow[j];
#pragma unroll
    for (int j = 0; j < 8; ++j) pa[j + 8] = prow[16 + j];
    __syncthreads();

    const v16h bv0 =
        *(const v16h*)(vhb + (size_t)it * 1024 + 0 * 512 + l * 16);
    const v16h bv1 =
        *(const v16h*)(vhb + (size_t)it * 1024 + 1 * 512 + l * 16);
    o0 = wmma_f16(pa, bv0, o0);
    o1 = wmma_f16(pa, bv1, o1);
  }

#pragma unroll
  for (int r = 0; r < 8; ++r) {
    int q = qt * 16 + hi * 8 + r;
    if (q < Q_LEN) {
      float invl = 1.0f / lrow[r];
      size_t base = (size_t)(q * BATCH + b) * D_MODEL + h * HEAD_DIM;
      att_out[base + (l & 15)] = o0[r] * invl;
      att_out[base + 16 + (l & 15)] = o1[r] * invl;
    }
  }
}

// ------------- Output projection + bias + residual ---------------------------
__global__ __launch_bounds__(32) void proj_out_kernel(
    const float* __restrict__ att, const float* __restrict__ Wo,
    const float* __restrict__ bo, const float* __restrict__ tgt,
    float* __restrict__ y) {
  const int rt = blockIdx.x, nt = blockIdx.y;
  const int l = threadIdx.x, m = l & 15, hi = l >> 4;
  const int row = rt * 16 + m;   // all 3200 rows valid
  const int ncol = nt * 16 + m;
  const float* arow = att + (size_t)row * D_MODEL;
  const float* wrow = Wo + (size_t)ncol * D_MODEL;
  v8f acc = {};
  for (int ks = 0; ks < D_MODEL; ks += 32) {
    v16h a = fragA_row(arow, ks, hi, 1.0f);
    v16h w = fragB_row(wrow, ks, hi);
    acc = wmma_f16(a, w, acc);
  }
#pragma unroll
  for (int r = 0; r < 8; ++r) {
    int orow = rt * 16 + hi * 8 + r;
    int col = nt * 16 + (l & 15);
    y[orow * D_MODEL + col] = acc[r] + bo[col] + tgt[orow * D_MODEL + col];
  }
}

// ------------- LayerNorm: one wave per row -----------------------------------
__global__ __launch_bounds__(32) void lnorm_kernel(
    const float* __restrict__ y, const float* __restrict__ g,
    const float* __restrict__ bb, float* __restrict__ out) {
  const int row = blockIdx.x;
  const int l = threadIdx.x;
  float v[8];
  float s = 0.0f;
#pragma unroll
  for (int i = 0; i < 8; i += 4) {
    float4 t = ld4(y + (size_t)row * D_MODEL + l * 8 + i);
    v[i + 0] = t.x; v[i + 1] = t.y; v[i + 2] = t.z; v[i + 3] = t.w;
    s += t.x + t.y + t.z + t.w;
  }
#pragma unroll
  for (int off = 1; off < 32; off <<= 1) s += __shfl_xor(s, off, 32);
  const float mu = s * (1.0f / 256.0f);
  float s2 = 0.0f;
#pragma unroll
  for (int i = 0; i < 8; ++i) {
    float d = v[i] - mu;
    s2 += d * d;
  }
#pragma unroll
  for (int off = 1; off < 32; off <<= 1) s2 += __shfl_xor(s2, off, 32);
  const float inv = rsqrtf(s2 * (1.0f / 256.0f) + 1e-5f);
#pragma unroll
  for (int i = 0; i < 8; ++i)
    out[(size_t)row * D_MODEL + l * 8 + i] =
        (v[i] - mu) * inv * g[l * 8 + i] + bb[l * 8 + i];
}

extern "C" void kernel_launch(void* const* d_in, const int* in_sizes, int n_in,
                              void* d_out, int out_size, void* d_ws,
                              size_t ws_size, hipStream_t stream) {
  (void)in_sizes; (void)n_in; (void)out_size; (void)ws_size;
  const float* tgt  = (const float*)d_in[0];
  const float* text = (const float*)d_in[1];
  const float* qpos = (const float*)d_in[2];
  const float* oov  = (const float*)d_in[3];
  const float* Wq = (const float*)d_in[4];
  const float* bq = (const float*)d_in[5];
  const float* Wk = (const float*)d_in[6];
  const float* bk = (const float*)d_in[7];
  const float* Wv = (const float*)d_in[8];
  const float* bv = (const float*)d_in[9];
  const float* Wo = (const float*)d_in[10];
  const float* bo = (const float*)d_in[11];
  const float* ln_g = (const float*)d_in[12];
  const float* ln_b = (const float*)d_in[13];
  const float* lsc  = (const float*)d_in[14];
  float* out = (float*)d_out;

  char* base = (char*)d_ws;
  size_t off = 0;
  auto carve = [&](size_t bytes) {
    void* p = base + off;
    off += (bytes + 255) & ~(size_t)255;
    return p;
  };
  float* kf = (float*)carve((size_t)T_PAD * D_MODEL * 4);
  float* vf = (float*)carve((size_t)T_PAD * D_MODEL * 4);
  float* qf = (float*)carve((size_t)BATCH * Q_PAD * D_MODEL * 4);
  _Float16* qh = (_Float16*)carve((size_t)BATCH * NHEAD * QT * 512 * 2);
  _Float16* kh = (_Float16*)carve((size_t)NHEAD * TT * 512 * 2);
  _Float16* vh = (_Float16*)carve((size_t)NHEAD * NIT * 2 * 512 * 2);
  float* att = (float*)carve((size_t)ROWS_QB * D_MODEL * 4);
  float* y   = (float*)carve((size_t)ROWS_QB * D_MODEL * 4);

  proj_text_kernel<<<dim3(TT, 16, 2), 32, 0, stream>>>(text, Wk, bk, Wv, bv,
                                                       kf, vf);
  proj_q_kernel<<<dim3(QT, 16, BATCH), 32, 0, stream>>>(tgt, qpos, Wq, bq, qf);
  pack_q_kernel<<<dim3(QT, NHEAD, BATCH), 32, 0, stream>>>(qf, qh);
  pack_k_kernel<<<dim3(TT, NHEAD), 32, 0, stream>>>(kf, kh);
  pack_v_kernel<<<dim3(NIT, 2, NHEAD), 32, 0, stream>>>(vf, vh);
  attn_kernel<<<dim3(QT, NHEAD, BATCH), 32, 0, stream>>>(qh, kh, vh, oov, lsc,
                                                         att);
  proj_out_kernel<<<dim3(200, 16), 32, 0, stream>>>(att, Wo, bo, tgt, y);
  lnorm_kernel<<<3200, 32, 0, stream>>>(y, ln_g, ln_b, out);
}
